// ModulatedConv2d_6786048327703
// MI455X (gfx1250) — matrix operand order repaired and checked by reading, not compiled, not run on hardware
//
#include <hip/hip_runtime.h>

// ---------------------------------------------------------------------------
// ModulatedConv2d for MI455X (gfx1250, wave32, WMMA).
//
// Roofline: conv = 77.3 GFLOP, min HBM traffic = 536 MB -> ~23 us @ 23.3 TB/s.
// bf16 V_WMMA_F32_16X16X32_BF16 matches that time scale, so the design is
// memory-first: one LDS-tiled NCHW->NHWC bf16 transpose (coalesced both
// sides), weights pre-packed into the exact A-fragment VGPR/lane layout, and
// activation halo tiles staged to LDS via GLOBAL_LOAD_ASYNC_TO_LDS_B128
// (ASYNCcnt) when the toolchain exposes the builtin.
// ---------------------------------------------------------------------------

typedef __attribute__((ext_vector_type(16))) __bf16        v16bf;
typedef __attribute__((ext_vector_type(8)))  float         v8f;
typedef __attribute__((ext_vector_type(8)))  unsigned int  v8u;
typedef __attribute__((ext_vector_type(4)))  unsigned int  v4u;

#define N_    16
#define CIN_  64
#define COUT_ 64
#define H_    256
#define W_    256
#define L_    512
#define KPOS_ 9
#define EPS_  1e-8f
#define AFRAG_PER_N (KPOS_ * 2 * 4 * 32 * 16)   // 36864 bf16 per sample

#if defined(__gfx1250__) && \
    __has_builtin(__builtin_amdgcn_global_load_async_to_lds_b128) && \
    __has_builtin(__builtin_amdgcn_s_wait_asynccnt)
#define USE_ASYNC_LDS 1
// The builtin's source operand is a generic pointer to a 16-byte int vector
// (per hipcc's diagnostic); the LDS destination is addrspace(3).
typedef int v4i_g __attribute__((vector_size(16)));
typedef __attribute__((address_space(3))) v4i_g* lds_v4p;
#else
#define USE_ASYNC_LDS 0
#endif

__device__ __forceinline__ unsigned short f32_to_bf16(float f) {
    unsigned int u = __float_as_uint(f);
    unsigned int r = u + 0x7fffu + ((u >> 16) & 1u);   // round-to-nearest-even
    return (unsigned short)(r >> 16);
}

__device__ __forceinline__ v16bf as_v16bf(v8u v) {
    union { v8u u; v16bf b; } c; c.u = v; return c.b;
}

// ---------------------------------------------------------------------------
// Kernel 0: x [N,CIN,H,W] f32 -> xT [N,H,W,CIN] bf16, LDS-tiled so both the
// global reads (256B-contiguous f32) and writes (128B-contiguous dwords of
// paired bf16) are coalesced. LDS pitch 65 (odd) -> bank-conflict-free.
// ---------------------------------------------------------------------------
__global__ __launch_bounds__(256)
void xpose_kernel(const float* __restrict__ x, unsigned short* __restrict__ xT) {
    const int n  = blockIdx.z;
    const int h  = blockIdx.y;
    const int w0 = blockIdx.x * 64;
    const int t  = threadIdx.x;

    __shared__ unsigned short lds[64 * 65];

#pragma unroll
    for (int i = 0; i < 16; ++i) {          // read: 64w consecutive per c row
        int lin = i * 256 + t;              // 0..4095
        int w = lin & 63;
        int c = lin >> 6;
        lds[c * 65 + w] = f32_to_bf16(x[((n * CIN_ + c) * H_ + h) * W_ + w0 + w]);
    }
    __syncthreads();
#pragma unroll
    for (int i = 0; i < 8; ++i) {           // write: 32 threads cover c=0..63
        int lin = i * 256 + t;              // 0..2047
        int w  = lin >> 5;                  // 0..63
        int c2 = (lin & 31) * 2;            // even c
        unsigned int lo = lds[c2 * 65 + w];
        unsigned int hi = lds[(c2 + 1) * 65 + w];
        *(unsigned int*)&xT[((n * H_ + h) * W_ + w0 + w) * CIN_ + c2] =
            lo | (hi << 16);
    }
}

// ---------------------------------------------------------------------------
// Kernel 1: style GEMM + modulate + demodulate + pack bf16 A-fragments.
// A-fragment layout (16-bit A 16x32, contiguous k, same order as the B loads):
//   lane<16 : M=lane,    K(i) = i
//   lane>=16: M=lane-16, K(i) = 16 + i
// Fragment index: ((kpos*2 + cinblk)*4 + coutTile)*512 + lane*16 + i
// ---------------------------------------------------------------------------
__global__ __launch_bounds__(256)
void wprep_kernel(const float* __restrict__ latent, const float* __restrict__ weight,
                  const float* __restrict__ style_w, const float* __restrict__ style_b,
                  unsigned short* __restrict__ wpack) {
    const int n = blockIdx.x, tid = threadIdx.x;
    __shared__ float style_s[CIN_];
    __shared__ float demod_s[COUT_];

    if (tid < CIN_) {                                   // style = latent @ W^T + b
        float acc = style_b[tid];
        const float* lat = latent + n * L_;
        const float* sw  = style_w + tid * L_;
        for (int l = 0; l < L_; ++l) acc += lat[l] * sw[l];
        style_s[tid] = acc;
    }
    __syncthreads();

    if (tid < COUT_) {                                  // demod = rsqrt(sum w_mod^2)
        float s = 0.f;
        const float* wrow = weight + tid * CIN_ * KPOS_;
        for (int cin = 0; cin < CIN_; ++cin) {
            float st = style_s[cin];
            for (int k = 0; k < KPOS_; ++k) {
                float v = wrow[cin * KPOS_ + k] * st;
                s += v * v;
            }
        }
        demod_s[tid] = rsqrtf(s + EPS_);
    }
    __syncthreads();

    for (int idx = tid; idx < AFRAG_PER_N; idx += 256) {
        int i    = idx & 15;
        int lane = (idx >> 4) & 31;
        int t    = (idx >> 9) & 3;
        int cb   = (idx >> 11) & 1;
        int kpos = idx >> 12;
        int m      = t * 16 + (lane & 15);
        int klocal = ((lane < 16) ? 0 : 16) + i;
        int cin    = cb * 32 + klocal;
        float v = weight[(m * CIN_ + cin) * KPOS_ + kpos] * style_s[cin] * demod_s[m];
        wpack[n * AFRAG_PER_N + idx] = f32_to_bf16(v);
    }
}

// ---------------------------------------------------------------------------
// Kernel 2: implicit-GEMM conv. Block = (n, row h, 128-pixel strip).
// 8 waves x 16 pixels; each wave owns all 64 couts (4 WMMA accumulators).
// K loop: 3x3 taps x 2 cin-blocks of 32 -> 18 steps x 4 WMMA.
// Halo tile staged to LDS via async-to-LDS (ASYNCcnt) when available.
// ---------------------------------------------------------------------------
__global__ __launch_bounds__(256)
void modconv_kernel(const unsigned short* __restrict__ xT,
                    const unsigned short* __restrict__ wpack,
                    float* __restrict__ out) {
    const int n   = blockIdx.z;
    const int h   = blockIdx.y;
    const int w0  = blockIdx.x * 128;
    const int tid = threadIdx.x;

    __shared__ __align__(32) unsigned short lds_x[3 * 130 * CIN_];   // 48.75 KB

    // Warm L2 for the next row strip (global_prefetch_b8, speculative).
    if (h + 2 < H_)
        __builtin_prefetch(&xT[(((n * H_ + h + 2) * W_ + w0 + (tid >> 1)) * CIN_)], 0, 1);

    // Stage halo tile: rows h-1..h+1, cols w0-1..w0+128, 64 cin, 16B chunks.
    for (int cidx = tid; cidx < 3 * 130 * 8; cidx += 256) {
        int c8  = (cidx & 7) * 8;                       // 8 bf16 = 16 bytes
        int px  = (cidx >> 3) % 130;
        int row = cidx / (130 * 8);
        int gh  = h - 1 + row;
        int gw  = w0 - 1 + px;
        unsigned short* lp = &lds_x[(row * 130 + px) * CIN_ + c8];
        if ((unsigned)gh < (unsigned)H_ && (unsigned)gw < (unsigned)W_) {
            const unsigned short* gp = &xT[(((n * H_ + gh) * W_ + gw) * CIN_) + c8];
#if USE_ASYNC_LDS
            __builtin_amdgcn_global_load_async_to_lds_b128(
                (v4i_g*)(void*)gp, (lds_v4p)(void*)lp, 0, 0);
#else
            *(v4u*)lp = *(const v4u*)gp;
#endif
        } else {
            v4u z = {};                                 // zero-pad OOB halo
            *(v4u*)lp = z;
        }
    }
#if USE_ASYNC_LDS
    __builtin_amdgcn_s_wait_asynccnt(0);
#endif
    __syncthreads();

    const int wave   = tid >> 5;
    const int lane   = tid & 31;
    const int lanelo = lane & 15;
    const int khalf  = (lane < 16) ? 0 : 16;            // B-fragment K offset

    v8f acc0 = {}, acc1 = {}, acc2 = {}, acc3 = {};
    const unsigned short* wp = wpack + n * AFRAG_PER_N;

    for (int kh = 0; kh < 3; ++kh)
        for (int kw = 0; kw < 3; ++kw)
#pragma unroll
            for (int cb = 0; cb < 2; ++cb) {
                int px = wave * 16 + lanelo + kw;       // tile starts at w0-1
                v16bf b = as_v16bf(
                    *(const v8u*)&lds_x[(kh * 130 + px) * CIN_ + cb * 32 + khalf]);
                const unsigned short* af = wp + (((kh * 3 + kw) * 2 + cb) << 11);
                v16bf a0 = as_v16bf(*(const v8u*)&af[0 * 512 + lane * 16]);
                v16bf a1 = as_v16bf(*(const v8u*)&af[1 * 512 + lane * 16]);
                v16bf a2 = as_v16bf(*(const v8u*)&af[2 * 512 + lane * 16]);
                v16bf a3 = as_v16bf(*(const v8u*)&af[3 * 512 + lane * 16]);
                acc0 = __builtin_amdgcn_wmma_f32_16x16x32_bf16(false, a0, false, b,
                                                               (short)0, acc0, false, false);
                acc1 = __builtin_amdgcn_wmma_f32_16x16x32_bf16(false, a1, false, b,
                                                               (short)0, acc1, false, false);
                acc2 = __builtin_amdgcn_wmma_f32_16x16x32_bf16(false, a2, false, b,
                                                               (short)0, acc2, false, false);
                acc3 = __builtin_amdgcn_wmma_f32_16x16x32_bf16(false, a3, false, b,
                                                               (short)0, acc3, false, false);
            }

    // C/D layout: lane<16 -> M=r, lane>=16 -> M=r+8; N = lanelo (pixel).
    // For fixed r, lanes 0..15 write 16 consecutive floats -> coalesced 64B.
    const int w    = w0 + wave * 16 + lanelo;
    const int mofs = (lane < 16) ? 0 : 8;
#pragma unroll
    for (int r = 0; r < 8; ++r) {
        int m = r + mofs;
        out[(((n * COUT_ + 0 * 16 + m) * H_ + h) * W_) + w] = acc0[r];
        out[(((n * COUT_ + 1 * 16 + m) * H_ + h) * W_) + w] = acc1[r];
        out[(((n * COUT_ + 2 * 16 + m) * H_ + h) * W_) + w] = acc2[r];
        out[(((n * COUT_ + 3 * 16 + m) * H_ + h) * W_) + w] = acc3[r];
    }
}

// ---------------------------------------------------------------------------
extern "C" void kernel_launch(void* const* d_in, const int* in_sizes, int n_in,
                              void* d_out, int out_size, void* d_ws, size_t ws_size,
                              hipStream_t stream) {
    const float* x       = (const float*)d_in[0];   // [N,CIN,H,W]
    const float* latent  = (const float*)d_in[1];   // [N,L]
    const float* weight  = (const float*)d_in[2];   // [COUT,CIN,3,3]
    const float* style_w = (const float*)d_in[3];   // [CIN,L]
    const float* style_b = (const float*)d_in[4];   // [CIN]
    float* out = (float*)d_out;                     // [N,COUT,H,W]

    unsigned short* xT    = (unsigned short*)d_ws;              // 134.2 MB bf16 NHWC
    unsigned short* wpack = xT + (size_t)N_ * H_ * W_ * CIN_;   // 1.18 MB A-fragments

    xpose_kernel<<<dim3(W_ / 64, H_, N_), 256, 0, stream>>>(x, xT);
    wprep_kernel<<<N_, 256, 0, stream>>>(latent, weight, style_w, style_b, wpack);
    modconv_kernel<<<dim3(W_ / 128, H_, N_), 256, 0, stream>>>(xT, wpack, out);
}
